// TransformerEncoderLayer_16801912062395
// MI455X (gfx1250) — matrix operand order repaired and checked by reading, not compiled, not run on hardware
//
#include <hip/hip_runtime.h>
#include <hip/hip_bf16.h>
#include <math.h>

#define S_ 128
#define B_ 8
#define E_ 256
#define H_ 8
#define D_ 32

typedef __attribute__((ext_vector_type(16))) __bf16 v16bf;
typedef __attribute__((ext_vector_type(8)))  float  v8f;
typedef __attribute__((ext_vector_type(4)))  float  v4f;

// ---------------------------------------------------------------------------
// P1: q/v projections of `query` (k is unused by the reference), head-ordered.
// ---------------------------------------------------------------------------
__global__ __launch_bounds__(256) void qv_proj_kernel(
    const float* __restrict__ query,   // (S,B,E)
    const float* __restrict__ W,       // (3E,E)
    const float* __restrict__ bias,    // (3E)
    float* __restrict__ qh,            // (B,H,S,D)
    float* __restrict__ vh)            // (B,H,S,D)
{
  const int sb = blockIdx.x;           // 0..S*B-1
  const int s = sb / B_, b = sb % B_;
  const int e = threadIdx.x;           // 0..255
  __shared__ float x[E_];
  x[e] = query[(size_t)(s * B_ + b) * E_ + e];
  __syncthreads();
  const float* wq = W + (size_t)e * E_;
  const float* wv = W + (size_t)(2 * E_ + e) * E_;
  float aq = bias[e];
  float av = bias[2 * E_ + e];
#pragma unroll 8
  for (int k = 0; k < E_; ++k) {
    aq = fmaf(x[k], wq[k], aq);
    av = fmaf(x[k], wv[k], av);
  }
  const int h = e / D_, d = e % D_;
  const size_t o = ((size_t)(b * H_ + h) * S_ + s) * D_ + d;
  qh[o] = aq;
  vh[o] = av;
}

// ---------------------------------------------------------------------------
// P2: qp = qh @ w1q + b1 (bias folded), vp = vh @ w1v.
// ---------------------------------------------------------------------------
__global__ __launch_bounds__(256) void qp_vp_kernel(
    const float* __restrict__ qh, const float* __restrict__ vh,
    const float* __restrict__ w1,   // (H, 3D, D)
    const float* __restrict__ b1,   // (H, D)
    float* __restrict__ qp, float* __restrict__ vp)
{
  const int t = blockIdx.x * blockDim.x + threadIdx.x;  // B*H*S*D threads
  const int e = t & (D_ - 1);
  const int h = (t >> 12) & (H_ - 1);
  const size_t base = (size_t)(t >> 5) * D_;            // (b,h,s) row base
  const float* wq = w1 + (size_t)h * 3 * D_ * D_ + e;   // w1q[d][e]
  const float* wv = wq + D_ * D_;                       // w1v[d][e]
  float aq = b1[h * D_ + e], av = 0.f;
#pragma unroll
  for (int d = 0; d < D_; ++d) {
    aq = fmaf(qh[base + d], wq[d * D_], aq);
    av = fmaf(vh[base + d], wv[d * D_], av);
  }
  qp[t] = aq;
  vp[t] = av;
}

// ---------------------------------------------------------------------------
// Main kernel: pp = pf @ w1p via v_wmma_f32_16x16x32_bf16, hidden add +
// leaky-relu (packed max/min form) + w2-reduction via an LDS transpose,
// masked softmax, attn @ vh.
// Grid: B*H*(S/16) blocks of 256 threads (8 waves). Each wave owns 2 l-rows.
// ---------------------------------------------------------------------------
#define PSTRIDE 20   // padded row stride of the reduction scratch (floats)

__global__ __launch_bounds__(256) void pair_attn_kernel(
    const float* __restrict__ pf,     // (B,S,S,E)
    const float* __restrict__ qp,     // (B,H,S,D)  (b1 pre-added)
    const float* __restrict__ vp,     // (B,H,S,D)
    const float* __restrict__ vh,     // (B,H,S,D)
    const float* __restrict__ w1,     // (H,3D,D)
    const float* __restrict__ w2,     // (H,D)
    const float* __restrict__ b2,     // (H)
    const unsigned char* __restrict__ mask, // (B,S) bool
    float* __restrict__ attn_out,     // (B,H,S,S)
    float* __restrict__ out_head)     // (B,H,S,D)
{
  const int blk = blockIdx.x;
  const int l0 = (blk & 7) * 16;
  const int h  = (blk >> 3) & 7;
  const int b  = blk >> 6;

  const int tid  = threadIdx.x;
  const int wave = tid >> 5;
  const int lane = tid & 31;

  __shared__ float s_vp[S_ * D_];          // 16 KB
  __shared__ float s_vh[S_ * D_];          // 16 KB
  __shared__ float s_qp[16 * D_];          //  2 KB
  __shared__ float s_w2[D_];
  __shared__ float s_maskadd[S_];          // 0 or -inf, additive
  __shared__ float s_logits[16 * S_];      //  8 KB
  __shared__ float s_part[8 * 16 * PSTRIDE]; // 10 KB reduction scratch

  {
    const size_t hb = (size_t)(b * H_ + h) * S_ * D_;
    for (int i = tid; i < S_ * D_; i += 256) {
      s_vp[i] = vp[hb + i];
      s_vh[i] = vh[hb + i];
    }
    const size_t qb = (size_t)((b * H_ + h) * S_ + l0) * D_;
    if (tid < 16 * D_) s_qp[tid] = qp[qb + tid];
    if (tid < D_) s_w2[tid] = w2[h * D_ + tid];
    if (tid < S_) s_maskadd[tid] = mask[b * S_ + tid] ? -INFINITY : 0.0f;
  }
  __syncthreads();

  // B fragment (w1p, 32x16 per N-half), bf16.  ISA 16-bit B layout:
  // lane%16 = column N, elements 0..7 -> K = kb+0..7, 8..15 -> K = kb+16..23, kb = 0/8.
  v16bf bfrag0, bfrag1;
  {
    const float* w1p = w1 + ((size_t)h * 3 * D_ + 2 * D_) * D_;  // w1p[k][n]
    const int kbase = (lane >= 16) ? 8 : 0;
    const int n = lane & 15;
#pragma unroll
    for (int e = 0; e < 16; ++e) {
      bfrag0[e] = (__bf16)w1p[(kbase + (e & 7) + (e >> 3) * 16) * D_ + n];
      bfrag1[e] = (__bf16)w1p[(kbase + (e & 7) + (e >> 3) * 16) * D_ + n + 16];
    }
  }
  const float b2v  = b2[h];
  const int   half = (lane >= 16) ? 1 : 0;
  const int   e0   = lane & 15;
  const float w2a  = s_w2[e0];
  const float w2b  = s_w2[e0 + 16];
  float* my_part = &s_part[wave * 16 * PSTRIDE];

  for (int li = 0; li < 2; ++li) {
    const int l_local = wave * 2 + li;
    const int l = l0 + l_local;
    const float qpa = s_qp[l_local * D_ + e0];
    const float qpb = s_qp[l_local * D_ + e0 + 16];
    const float* prow = pf + ((size_t)(b * S_ + l) * S_) * E_ + h * D_;

    for (int wt = 0; wt < 8; ++wt) {
      const int w0 = wt * 16;
      // A fragment 16x32 bf16: lane%16 = row M (w index), K mapping:
      // elements 0..7 -> K = kb+0..7, elements 8..15 -> K = kb+16..23,
      // kb = 0 (lanes 0-15) or 8 (lanes 16-31).  pf is streamed exactly
      // once -> non-temporal loads keep it out of the way in cache.
      const float* arow = prow + (size_t)(w0 + e0) * E_ + half * 8;
      v4f r0 = __builtin_nontemporal_load((const v4f*)(arow + 0));
      v4f r1 = __builtin_nontemporal_load((const v4f*)(arow + 4));
      v4f r2 = __builtin_nontemporal_load((const v4f*)(arow + 16));
      v4f r3 = __builtin_nontemporal_load((const v4f*)(arow + 20));
      if (wt < 7) __builtin_prefetch(arow + 16 * E_, 0, 3);
      v16bf a;
      a[0]  = (__bf16)r0.x; a[1]  = (__bf16)r0.y; a[2]  = (__bf16)r0.z; a[3]  = (__bf16)r0.w;
      a[4]  = (__bf16)r1.x; a[5]  = (__bf16)r1.y; a[6]  = (__bf16)r1.z; a[7]  = (__bf16)r1.w;
      a[8]  = (__bf16)r2.x; a[9]  = (__bf16)r2.y; a[10] = (__bf16)r2.z; a[11] = (__bf16)r2.w;
      a[12] = (__bf16)r3.x; a[13] = (__bf16)r3.y; a[14] = (__bf16)r3.z; a[15] = (__bf16)r3.w;

      v8f c0 = {0.f, 0.f, 0.f, 0.f, 0.f, 0.f, 0.f, 0.f};
      v8f c1 = c0;
      c0 = __builtin_amdgcn_wmma_f32_16x16x32_bf16(false, a, false, bfrag0,
                                                   (short)0, c0, false, false);
      c1 = __builtin_amdgcn_wmma_f32_16x16x32_bf16(false, a, false, bfrag1,
                                                   (short)0, c1, false, false);

      // C layout: VGPR r -> M = r (lanes 0-15) / r+8 (lanes 16-31), N = lane%16.
      // leaky(x) = max(x,0) + 0.01*min(x,0) stays in packed pk_max/pk_min/pk_fma
      // form (vs cmp+cndmask per scalar).
#pragma unroll
      for (int r = 0; r < 8; ++r) {
        const int w = w0 + r + half * 8;
        float x0 = c0[r] + qpa + s_vp[w * D_ + e0];
        float x1 = c1[r] + qpb + s_vp[w * D_ + e0 + 16];
        float h0 = fmaxf(x0, 0.f) + 0.01f * fminf(x0, 0.f);
        float h1 = fmaxf(x1, 0.f) + 0.01f * fminf(x1, 0.f);
        my_part[(r + half * 8) * PSTRIDE + e0] = fmaf(h0, w2a, h1 * w2b);
      }
      // Transpose-reduce: lane i (i<16) sums row i (16 floats, 4x b128 LDS
      // loads, bank-conflict-free at 80B stride).  Same-wave DS ops are
      // processed in order, so no barrier is needed.
      if (lane < 16) {
        const v4f* rp = (const v4f*)&my_part[lane * PSTRIDE];
        v4f a0 = rp[0], a1 = rp[1], a2 = rp[2], a3 = rp[3];
        float s = ((a0.x + a0.y) + (a0.z + a0.w)) + ((a1.x + a1.y) + (a1.z + a1.w))
                + ((a2.x + a2.y) + (a2.z + a2.w)) + ((a3.x + a3.y) + (a3.z + a3.w));
        const int w = w0 + lane;
        s_logits[l_local * S_ + w] = s + b2v + s_maskadd[w];
      }
    }
  }
  __syncthreads();

  // Softmax over w (wave-local: each wave owns its 2 rows) then attn @ vh.
  for (int li = 0; li < 2; ++li) {
    const int l_local = wave * 2 + li;
    const int l = l0 + l_local;
    float* lg = &s_logits[l_local * S_];
    float v0 = lg[lane], v1 = lg[lane + 32], v2 = lg[lane + 64], v3 = lg[lane + 96];
    float mx = fmaxf(fmaxf(v0, v1), fmaxf(v2, v3));
    for (int o = 1; o < 32; o <<= 1) mx = fmaxf(mx, __shfl_xor(mx, o, 32));
    float e0_ = __expf(v0 - mx), e1_ = __expf(v1 - mx);
    float e2_ = __expf(v2 - mx), e3_ = __expf(v3 - mx);
    float sum = e0_ + e1_ + e2_ + e3_;
    for (int o = 1; o < 32; o <<= 1) sum += __shfl_xor(sum, o, 32);
    const float inv = 1.0f / sum;
    e0_ *= inv; e1_ *= inv; e2_ *= inv; e3_ *= inv;
    float* ao = attn_out + ((size_t)(b * H_ + h) * S_ + l) * S_;
    ao[lane]      = e0_;  ao[lane + 32] = e1_;
    ao[lane + 64] = e2_;  ao[lane + 96] = e3_;
    lg[lane]      = e0_;  lg[lane + 32] = e1_;
    lg[lane + 64] = e2_;  lg[lane + 96] = e3_;
    __syncthreads();
    // out[l, d=lane] = sum_w attn[l,w] * vh[w,d]
    float acc = 0.f;
    for (int w = 0; w < S_; ++w) acc = fmaf(lg[w], s_vh[w * D_ + lane], acc);
    out_head[((size_t)(b * H_ + h) * S_ + l) * D_ + lane] = acc;
  }
}

// ---------------------------------------------------------------------------
// Final projection: out(s,b,:) = out_head(s,b,:) @ Wout^T + bias.
// ---------------------------------------------------------------------------
__global__ __launch_bounds__(256) void out_proj_kernel(
    const float* __restrict__ out_head, // (B,H,S,D)
    const float* __restrict__ W,        // (E,E)
    const float* __restrict__ bias,     // (E)
    float* __restrict__ out)            // (S,B,E)
{
  const int sb = blockIdx.x;
  const int s = sb / B_, b = sb % B_;
  const int e = threadIdx.x;
  __shared__ float x[E_];
  const int h = e / D_, d = e % D_;
  x[e] = out_head[((size_t)(b * H_ + h) * S_ + s) * D_ + d];
  __syncthreads();
  const float* w = W + (size_t)e * E_;
  float a = bias[e];
#pragma unroll 8
  for (int k = 0; k < E_; ++k) a = fmaf(x[k], w[k], a);
  out[(size_t)(s * B_ + b) * E_ + e] = a;
}

// ---------------------------------------------------------------------------
extern "C" void kernel_launch(void* const* d_in, const int* in_sizes, int n_in,
                              void* d_out, int out_size, void* d_ws, size_t ws_size,
                              hipStream_t stream) {
  const float*         query = (const float*)d_in[0];
  // d_in[1] (key) and d_in[2] (value) are unused by the reference math.
  const unsigned char* mask  = (const unsigned char*)d_in[3]; // bool (B,S)
  const float*         pf    = (const float*)d_in[4];
  const float*         inW   = (const float*)d_in[5];
  const float*         inB   = (const float*)d_in[6];
  const float*         w1    = (const float*)d_in[7];
  const float*         b1    = (const float*)d_in[8];
  const float*         w2    = (const float*)d_in[9];
  const float*         b2    = (const float*)d_in[10];
  const float*         outW  = (const float*)d_in[11];
  const float*         outB  = (const float*)d_in[12];

  float* ws = (float*)d_ws;
  const size_t N = (size_t)B_ * H_ * S_ * D_;  // 262144
  float* qh = ws;
  float* vh = ws + N;
  float* qp = ws + 2 * N;
  float* vp = ws + 3 * N;
  float* oh = ws + 4 * N;

  float* out  = (float*)d_out;
  float* attn = out + (size_t)S_ * B_ * E_;

  qv_proj_kernel<<<S_ * B_, 256, 0, stream>>>(query, inW, inB, qh, vh);
  qp_vp_kernel<<<(int)(N / 256), 256, 0, stream>>>(qh, vh, w1, b1, qp, vp);
  pair_attn_kernel<<<B_ * H_ * (S_ / 16), 256, 0, stream>>>(
      pf, qp, vp, vh, w1, w2, b2, mask, attn, oh);
  out_proj_kernel<<<S_ * B_, 256, 0, stream>>>(oh, outW, outB, out);
}
